// mGRU_29145648070818
// MI455X (gfx1250) — matrix operand order repaired and checked by reading, not compiled, not run on hardware
//
#include <hip/hip_runtime.h>
#include <hip/hip_bf16.h>

// ---------------------------------------------------------------------------
// mGRU recurrence for MI455X (gfx1250, wave32, WMMA bf16 -> f32 accum)
//   B=8, T=24, N=4096, Fi=64, Hh=128, C=192
// One workgroup = 1 batch x 128 columns, runs all T steps (h stays on chip).
// 8 waves/WG, each wave owns 16 columns (one WMMA N-tile).
// Biases live in LDS and are folded into the WMMA accumulator initialization.
// A-fragments use a 3-deep rotation to keep ds loads ahead of the WMMA pipe.
// Activations use v_tanh_f32 (if available) / v_exp+v_rcp fast paths.
// ---------------------------------------------------------------------------

typedef __bf16 bf16_t;
typedef bf16_t bf16x8 __attribute__((ext_vector_type(8)));
typedef bf16_t v16bf  __attribute__((ext_vector_type(16)));
typedef float  v8f    __attribute__((ext_vector_type(8)));
typedef float  f4     __attribute__((ext_vector_type(4)));

#define DEV __device__ __forceinline__

// Problem sizes
constexpr int Bb = 8, Tt = 24, Nn = 4096, FI = 64, HH = 128, CC = 192;
constexpr int NT = 128;          // columns per workgroup (8 waves x 16)

// LDS layout (element strides padded for 16B/32B vector alignment + bank spread)
constexpr int WG_S  = 200;       // Wg row stride (bf16 entries), 400B = 25*16
constexpr int WC_S  = 200;       // Wc row stride
constexpr int WO_S  = 136;       // Wo row stride, 272B = 17*16
constexpr int HX_S  = 144;       // h/rh bounce buf column stride (288B = 9*32)
constexpr int OUT_S = 72;        // out staging column stride (288B, f32)

constexpr int OFF_WC   = 256 * WG_S * 2;                 // 102400
constexpr int OFF_WO   = OFF_WC + 128 * WC_S * 2;        // 153600
constexpr int OFF_HX   = OFF_WO + 64  * WO_S * 2;        // 171008 (32B aligned)
constexpr int OFF_OUT  = OFF_HX + NT  * HX_S * 2;        // 207872 (32B aligned)
constexpr int OFF_BIAS = OFF_OUT + NT * OUT_S * 4;       // 244736 (32B aligned)
constexpr int NBIAS    = 256 + 128 + 64;                 // gate_b | cand_b | out_b
constexpr int SMEM_BYTES = OFF_BIAS + NBIAS * 4;         // 246528 < 320KB WGP LDS

DEV v8f wmma_bf16(v16bf a, v16bf b, v8f c) {
    // D = A(16x32) * B(32x16) + C, f32 accumulate
    return __builtin_amdgcn_wmma_f32_16x16x32_bf16(
        /*neg_a=*/false, a, /*neg_b=*/false, b,
        /*c_mod=*/(short)0, c, /*reuse_a=*/false, /*reuse_b=*/false);
}

// A fragment (16x32 bf16) from LDS weight matrix, row-major [o][c].
// K order used here (matches B): lane<16 holds K=0..15; lane>=16 holds K=16..31.
DEV v16bf load_a(const bf16_t* W, int stride, int obase, int kk, int nlo, int hi8) {
    const bf16_t* p = W + (obase + nlo) * stride + kk * 32 + hi8 * 2;
    bf16x8 lo = *(const bf16x8*)(p);        // 16B, aligned by stride choice
    bf16x8 hh = *(const bf16x8*)(p + 8);
    return __builtin_shufflevector(lo, hh, 0,1,2,3,4,5,6,7,8,9,10,11,12,13,14,15);
}

// B fragment (32x16 bf16) from LDS bounce buffer, layout [n][c] (c contiguous).
DEV v16bf load_bh(const bf16_t* hxw, int kk, int nlo, int hi) {
    return *(const v16bf*)(hxw + nlo * HX_S + kk * 32 + hi * 16);   // 32B ds load
}

// B fragment for x rows: straight from global f32 (f contiguous per n) -> bf16.
DEV v16bf load_bx(const float* xg, int kk, int nlo, int hi) {
    const float* p = xg + nlo * FI + kk * 32 + hi * 16;
    f4 q0 = *(const f4*)(p);
    f4 q1 = *(const f4*)(p + 4);
    f4 q2 = *(const f4*)(p + 8);
    f4 q3 = *(const f4*)(p + 12);
    f4 q[4] = {q0, q1, q2, q3};
    v16bf r;
#pragma unroll
    for (int i = 0; i < 16; ++i) r[i] = (bf16_t)q[i >> 2][i & 3];
    return r;
}

// C/D f32 fragment -> LDS bf16 [n][c]; entries are contiguous per lane, so pack
// into one 16B ds_store_b128 (alignment guaranteed by HX_S and obase multiples).
DEV void store_cd(bf16_t* hxw, int obase, v8f f, int nlo, int hi8) {
    bf16x8 vv;
#pragma unroll
    for (int v = 0; v < 8; ++v) vv[v] = (bf16_t)f[v];
    *(bf16x8*)(hxw + nlo * HX_S + obase + hi8) = vv;
}

// 6-step K GEMM for one 16-row M tile; bias folded into accumulator init;
// 3-deep A-fragment rotation keeps three ds load pairs in flight.
DEV v8f mm6(const bf16_t* W, int stride, int obase, const float* biasLds,
            v16bf b0, v16bf b1, v16bf b2, v16bf b3, v16bf b4, v16bf b5,
            int nlo, int hi8) {
    v8f acc = *(const v8f*)(biasLds + obase + hi8);   // bias as C (rows = M)
    v16bf a0 = load_a(W, stride, obase, 0, nlo, hi8);
    v16bf a1 = load_a(W, stride, obase, 1, nlo, hi8);
    v16bf a2 = load_a(W, stride, obase, 2, nlo, hi8);
    acc = wmma_bf16(a0, b0, acc);
    a0 = load_a(W, stride, obase, 3, nlo, hi8);
    acc = wmma_bf16(a1, b1, acc);
    a1 = load_a(W, stride, obase, 4, nlo, hi8);
    acc = wmma_bf16(a2, b2, acc);
    a2 = load_a(W, stride, obase, 5, nlo, hi8);
    acc = wmma_bf16(a0, b3, acc);
    acc = wmma_bf16(a1, b4, acc);
    acc = wmma_bf16(a2, b5, acc);
    return acc;
}

// 4-step K GEMM (output projection)
DEV v8f mm4(const bf16_t* W, int stride, int obase, const float* biasLds,
            v16bf b0, v16bf b1, v16bf b2, v16bf b3, int nlo, int hi8) {
    v8f acc = *(const v8f*)(biasLds + obase + hi8);
    v16bf a0 = load_a(W, stride, obase, 0, nlo, hi8);
    v16bf a1 = load_a(W, stride, obase, 1, nlo, hi8);
    v16bf a2 = load_a(W, stride, obase, 2, nlo, hi8);
    acc = wmma_bf16(a0, b0, acc);
    a0 = load_a(W, stride, obase, 3, nlo, hi8);
    acc = wmma_bf16(a1, b1, acc);
    acc = wmma_bf16(a2, b2, acc);
    acc = wmma_bf16(a0, b3, acc);
    return acc;
}

// Fast activations: single-instruction v_tanh_f32 when the builtin exists,
// else v_exp_f32 + v_rcp_f32 (no IEEE divide chains).
DEV float tanh1(float x) {
#if __has_builtin(__builtin_amdgcn_tanhf)
    return __builtin_amdgcn_tanhf(x);
#elif __has_builtin(__builtin_amdgcn_tanh_f32)
    return __builtin_amdgcn_tanh_f32(x);
#else
    float e = __expf(-2.0f * x);
    return (1.0f - e) * __builtin_amdgcn_rcpf(1.0f + e);
#endif
}
DEV float sigf(float x) { return fmaf(tanh1(x * 0.5f), 0.5f, 0.5f); }

__global__ __launch_bounds__(256, 1)
void mgru_kernel(const float* __restrict__ x,
                 const float* __restrict__ gw, const float* __restrict__ gb,
                 const float* __restrict__ cw, const float* __restrict__ cb,
                 const float* __restrict__ ow, const float* __restrict__ ob,
                 float* __restrict__ out) {
    extern __shared__ __align__(32) char smem[];
    bf16_t* sWg = (bf16_t*)(smem);
    bf16_t* sWc = (bf16_t*)(smem + OFF_WC);
    bf16_t* sWo = (bf16_t*)(smem + OFF_WO);
    bf16_t* sHx = (bf16_t*)(smem + OFF_HX);
    float*  sOut  = (float*)(smem + OFF_OUT);
    float*  sBias = (float*)(smem + OFF_BIAS);

    const int tid = threadIdx.x;
    const int lane = tid & 31, wave = tid >> 5;
    const int nlo = lane & 15, hi = lane >> 4, hi8 = hi * 8;
    const int ntile = blockIdx.x, b = blockIdx.y;

    // ---- One-time weight conversion f32 -> bf16 into LDS (weights stationary).
    for (int i = tid; i < 256 * CC; i += 256) {
        int o = i / CC, c = i - o * CC;
        sWg[o * WG_S + c] = (bf16_t)gw[(o * CC + c) * 2 + 1];
    }
    for (int i = tid; i < 128 * CC; i += 256) {
        int o = i / CC, c = i - o * CC;
        sWc[o * WC_S + c] = (bf16_t)cw[(o * CC + c) * 2 + 1];
    }
    for (int i = tid; i < 64 * 128; i += 256) {
        int o = i >> 7, c = i & 127;
        sWo[o * WO_S + c] = (bf16_t)ow[i];
    }
    for (int i = tid; i < NBIAS; i += 256) {
        float v;
        if (i < 256)      v = gb[i];
        else if (i < 384) v = cb[i - 256];
        else              v = ob[i - 384];
        sBias[i] = v;
    }
    __syncthreads();

    const float* gBias = sBias;          // 256 entries
    const float* cBias = sBias + 256;    // 128
    const float* oBias = sBias + 384;    // 64

    bf16_t* hx_w  = sHx  + (wave * 16) * HX_S;   // this wave's 16 columns
    float*  out_w = sOut + (wave * 16) * OUT_S;

    // Recurrent state: h (128 x 16) in 8 f32 C/D fragments + bf16 B-layout copies.
    v8f   hF[8];
    v16bf hB[4];
    v8f   zF[8];
#pragma unroll
    for (int m = 0; m < 8; ++m) hF[m] = {};
#pragma unroll
    for (int k = 0; k < 4; ++k) hB[k] = {};

    const int n0 = ntile * NT + wave * 16;
    const float* xg0 = x + ((size_t)b * Tt * Nn + n0) * FI;

    for (int t = 0; t < Tt; ++t) {
        const float* xg = xg0 + (size_t)t * Nn * FI;

        // x rows of the B operand (c = 0..63), direct from HBM, single touch.
        v16bf xB0 = load_bx(xg, 0, nlo, hi);
        v16bf xB1 = load_bx(xg, 1, nlo, hi);
        if (t + 1 < Tt)  // warm L2/L0 for next timestep's tile (global_prefetch_b8)
            __builtin_prefetch((const char*)(xg + (size_t)Nn * FI) + lane * 128, 0, 1);

        // ---- r gates (Wg rows 0..127): sigmoid, fold into r*h, stash as bf16.
#pragma unroll
        for (int m = 0; m < 8; ++m) {
            v8f acc = mm6(sWg, WG_S, m * 16, gBias,
                          xB0, xB1, hB[0], hB[1], hB[2], hB[3], nlo, hi8);
            v8f rh;
#pragma unroll
            for (int v = 0; v < 8; ++v) rh[v] = sigf(acc[v]) * hF[m][v];
            store_cd(hx_w, m * 16, rh, nlo, hi8);
        }

        // ---- z gates (Wg rows 128..255): keep in registers for the blend.
#pragma unroll
        for (int m = 0; m < 8; ++m) {
            v8f acc = mm6(sWg, WG_S, 128 + m * 16, gBias,
                          xB0, xB1, hB[0], hB[1], hB[2], hB[3], nlo, hi8);
#pragma unroll
            for (int v = 0; v < 8; ++v) zF[m][v] = sigf(acc[v]);
        }

        // ---- candidate: cinp = [x ; r*h]; h = h + z*(tanh(.) - h)
        {
            v16bf c2 = load_bh(hx_w, 0, nlo, hi);
            v16bf c3 = load_bh(hx_w, 1, nlo, hi);
            v16bf c4 = load_bh(hx_w, 2, nlo, hi);
            v16bf c5 = load_bh(hx_w, 3, nlo, hi);
#pragma unroll
            for (int m = 0; m < 8; ++m) {
                v8f acc = mm6(sWc, WC_S, m * 16, cBias,
                              xB0, xB1, c2, c3, c4, c5, nlo, hi8);
#pragma unroll
                for (int v = 0; v < 8; ++v) {
                    float ht = tanh1(acc[v]);
                    float z = zF[m][v];
                    hF[m][v] = hF[m][v] + z * (ht - hF[m][v]);
                }
            }
        }

        // ---- refresh bf16 B-layout h (for out-proj now, gate GEMM next step)
#pragma unroll
        for (int m = 0; m < 8; ++m) store_cd(hx_w, m * 16, hF[m], nlo, hi8);
#pragma unroll
        for (int k = 0; k < 4; ++k) hB[k] = load_bh(hx_w, k, nlo, hi);

        // ---- output projection: relu(Wo(64x128) @ h + ob) -> LDS staging
#pragma unroll
        for (int m = 0; m < 4; ++m) {
            v8f acc = mm4(sWo, WO_S, m * 16, oBias,
                          hB[0], hB[1], hB[2], hB[3], nlo, hi8);
#pragma unroll
            for (int v = 0; v < 8; ++v)
                out_w[nlo * OUT_S + m * 16 + hi8 + v] = fmaxf(acc[v], 0.0f);
        }
        __syncthreads();

        // ---- coalesced store: out[b][t][n][o], 16B per lane, contiguous
        float* og = out + (((size_t)b * Tt + t) * Nn + (size_t)ntile * NT) * FI;
#pragma unroll
        for (int i = 0; i < (NT * FI / 4) / 256; ++i) {
            int idx = i * 256 + tid;
            int n = idx >> 4, q = idx & 15;
            *(f4*)(og + (size_t)n * FI + q * 4) = *(const f4*)(sOut + n * OUT_S + q * 4);
        }
        __syncthreads();
    }
}

extern "C" void kernel_launch(void* const* d_in, const int* in_sizes, int n_in,
                              void* d_out, int out_size, void* d_ws, size_t ws_size,
                              hipStream_t stream) {
    const float* x  = (const float*)d_in[0];
    const float* gw = (const float*)d_in[1];
    const float* gb = (const float*)d_in[2];
    const float* cw = (const float*)d_in[3];
    const float* cb = (const float*)d_in[4];
    const float* ow = (const float*)d_in[5];
    const float* ob = (const float*)d_in[6];
    float* out = (float*)d_out;

    dim3 grid(Nn / NT, Bb);   // 32 x 8 workgroups
    mgru_kernel<<<grid, 256, SMEM_BYTES, stream>>>(x, gw, gb, cw, cb, ow, ob, out);
}
